// leNet_73409581023397
// MI455X (gfx1250) — compile-verified, matrix operand order checked by reading, not run on hardware
//
#include <hip/hip_runtime.h>

typedef __attribute__((ext_vector_type(16))) _Float16 v16h;
typedef __attribute__((ext_vector_type(8)))  _Float16 v8h;
typedef __attribute__((ext_vector_type(8)))  float    v8f;
typedef __attribute__((ext_vector_type(4)))  int      v4i;

// ---- optional CDNA5 async global->LDS path (ASYNCcnt), with safe fallback ----
#if defined(__has_builtin)
#  if __has_builtin(__builtin_amdgcn_global_load_async_to_lds_b128) && \
      __has_builtin(__builtin_amdgcn_s_wait_asynccnt)
#    define USE_ASYNC_LDS 1
#  endif
#endif
#ifndef USE_ASYNC_LDS
#  define USE_ASYNC_LDS 0
#endif

__device__ __forceinline__ void copy16_g2l(_Float16* dst_lds, const _Float16* src_glob) {
#if USE_ASYNC_LDS
  __builtin_amdgcn_global_load_async_to_lds_b128(
      (__attribute__((address_space(1))) v4i*)(unsigned long long)(const void*)src_glob,
      (__attribute__((address_space(3))) v4i*)dst_lds, 0, 0);
#else
  *(v8h*)dst_lds = *(const v8h*)src_glob;
#endif
}

__device__ __forceinline__ void async_join() {
#if USE_ASYNC_LDS
  __builtin_amdgcn_s_wait_asynccnt(0);
#endif
}

// C3 sparse connectivity: C3_POS[oc][ic] = branch-input position j, or -1.
__device__ __constant__ int C3_POS[16][6] = {
  { 0, 1, 2,-1,-1,-1}, {-1, 0, 1, 2,-1,-1}, {-1,-1, 0, 1, 2,-1}, {-1,-1,-1, 0, 1, 2},
  { 2,-1,-1,-1, 0, 1}, { 1, 2,-1,-1,-1, 0},
  { 0, 1, 2, 3,-1,-1}, {-1, 0, 1, 2, 3,-1}, {-1,-1, 0, 1, 2, 3}, { 3,-1,-1, 0, 1, 2},
  { 2, 3,-1,-1, 0, 1}, { 1, 2, 3,-1,-1, 0},
  { 0, 1,-1, 2, 3,-1}, {-1, 0, 1,-1, 2, 3}, { 3,-1, 0, 1,-1, 2},
  { 0, 1, 2, 3, 4, 5}
};

__device__ __forceinline__ float lenet_act(float v) {
  return 1.7159f * tanhf(0.66666667f * v);
}

// K index held by (hi, i) slot of a 16-bit WMMA A/B fragment (16x32 / 32x16).
__device__ __forceinline__ constexpr int wmma_kidx(int hi, int i) {
  return (i < 8) ? (hi * 8 + i) : (16 + hi * 8 + (i - 8));
}

// Build a 16-half fragment from two aligned 8-half runs (2x b128 + shuffle).
__device__ __forceinline__ v16h frag16(const _Float16* p0, const _Float16* p1) {
  v8h lo = *(const v8h*)p0;
  v8h hi = *(const v8h*)p1;
  return __builtin_shufflevector(lo, hi, 0,1,2,3,4,5,6,7,8,9,10,11,12,13,14,15);
}

// ---------------------------------------------------------------------------
// Prep: padded f16 GEMM B-operands.
//   W3 : [16][160]  dense-scattered C3 kernel, W5 : [128][416], W6 : [96][128]
// ---------------------------------------------------------------------------
__global__ void prep_weights(const float* __restrict__ w3, const float* __restrict__ w4,
                             const float* __restrict__ w6, const float* __restrict__ c5w,
                             const float* __restrict__ f6w,
                             _Float16* __restrict__ W3, _Float16* __restrict__ W5,
                             _Float16* __restrict__ W6) {
  int id = blockIdx.x * blockDim.x + threadIdx.x;
  const int N3 = 16 * 160, N5 = 128 * 416, N6 = 96 * 128;
  if (id < N3) {
    int oc = id / 160, k = id - oc * 160;
    float v = 0.f;
    if (k < 150) {
      int ic = k / 25, tap = k - ic * 25;
      int pos = C3_POS[oc][ic];
      if (pos >= 0) {
        if (oc < 6)       v = w3[(oc * 3 + pos) * 25 + tap];
        else if (oc < 15) v = w4[((oc - 6) * 4 + pos) * 25 + tap];
        else              v = w6[pos * 25 + tap];
      }
    }
    W3[id] = (_Float16)v;
  } else if (id < N3 + N5) {
    int e = id - N3;
    int oc = e / 416, k = e - oc * 416;
    float v = (oc < 120 && k < 400) ? c5w[oc * 400 + k] : 0.f;
    W5[e] = (_Float16)v;
  } else if (id < N3 + N5 + N6) {
    int e = id - N3 - N5;
    int n = e / 128, k = e - n * 128;
    float v = (n < 84 && k < 120) ? f6w[n * 120 + k] : 0.f;
    W6[e] = (_Float16)v;
  }
}

// ---------------------------------------------------------------------------
// C1 (1->6, 5x5) fused with S2 sumpool*scale+bias + act. h1: f16 [B][6][14][14]
// ---------------------------------------------------------------------------
__global__ void c1_s2_act(const float* __restrict__ x, const float* __restrict__ c1w,
                          const float* __restrict__ c1b, const float* __restrict__ s2w,
                          const float* __restrict__ s2b, _Float16* __restrict__ h1,
                          int B) {
  int id = blockIdx.x * blockDim.x + threadIdx.x;
  if (id >= B * 196) return;
  int b = id / 196, p = id - b * 196;
  int oy = p / 14, ox = p - oy * 14;
  float patch[6][6];
  const float* xb = x + b * 1024 + (2 * oy) * 32 + 2 * ox;
  __builtin_prefetch(xb + 1024, 0, 0);   // global_prefetch_b8: next image row group
  #pragma unroll
  for (int i = 0; i < 6; ++i)
    #pragma unroll
    for (int j = 0; j < 6; ++j)
      patch[i][j] = xb[i * 32 + j];
  #pragma unroll
  for (int oc = 0; oc < 6; ++oc) {
    const float* w = c1w + oc * 25;
    float acc = 0.f;
    #pragma unroll
    for (int py = 0; py < 2; ++py)
      #pragma unroll
      for (int px = 0; px < 2; ++px)
        #pragma unroll
        for (int ky = 0; ky < 5; ++ky)
          #pragma unroll
          for (int kx = 0; kx < 5; ++kx)
            acc = fmaf(patch[py + ky][px + kx], w[ky * 5 + kx], acc);
    acc += 4.f * c1b[oc];
    float v = acc * s2w[oc] + s2b[oc];
    h1[(b * 6 + oc) * 196 + oy * 14 + ox] = (_Float16)lenet_act(v);
  }
}

// ---------------------------------------------------------------------------
// C3 im2col GEMM on WMMA: M = B*100, N = 16, K = 160 (150 valid).
// Block = 8 waves = 128 rows, touching <=3 images; their h1 slabs are staged
// in LDS (async) once. A-fragments gather from LDS with compile-time im2col
// offsets; row->(b,p) decomposition uses one uniform divide + compares.
// ---------------------------------------------------------------------------
template <int HI>
__device__ __forceinline__ void c3_build_a(v16h (&afr)[5], const _Float16* basep) {
  #pragma unroll
  for (int kt = 0; kt < 5; ++kt) {
    #pragma unroll
    for (int i = 0; i < 16; ++i) {
      const int k = kt * 32 + wmma_kidx(HI, i);
      if (k < 150) {
        const int ic = k / 25, q = k % 25;
        const int off = ic * 196 + (q / 5) * 14 + (q % 5);  // compile-time
        afr[kt][i] = basep[off];
      } else {
        afr[kt][i] = (_Float16)0.f;
      }
    }
  }
}

__global__ void c3_wmma(const _Float16* __restrict__ h1, const _Float16* __restrict__ W3,
                        const float* __restrict__ c3b, float* __restrict__ c3o,
                        int Mtotal, int B) {
  __shared__ alignas(16) _Float16 sh1[3 * 1176];
  int m_base = blockIdx.x * 128;
  int b0 = m_base / 100;
  // cooperative stage: 441 chunks of 8 halves
  for (int c = threadIdx.x; c < 3 * 147; c += blockDim.x) {
    int j = c / 147;
    if (b0 + j < B)
      copy16_g2l(&sh1[c * 8], h1 + (size_t)(b0 + j) * 1176 + (size_t)(c - j * 147) * 8);
  }
  async_join();
  __syncthreads();

  int warp = threadIdx.x >> 5;
  int lane = threadIdx.x & 31;
  int m0 = m_base + warp * 16;
  if (m0 >= Mtotal) return;               // wave-uniform
  int hi = lane >> 4, rlo = lane & 15;

  // wave-uniform divide once; per-lane rows resolved with compares
  int b_first = m0 / 100;
  int p0 = m0 - b_first * 100;            // 0..99
  int pa = p0 + rlo;                      // < 115
  int carry_a = (pa >= 100) ? 1 : 0;
  int b  = b_first + carry_a;
  int p  = pa - carry_a * 100;
  int oy = p / 10, ox = p - oy * 10;
  const _Float16* basep = sh1 + (b - b0) * 1176 + oy * 14 + ox;

  v16h afr[5];
  if (hi == 0) c3_build_a<0>(afr, basep);
  else         c3_build_a<1>(afr, basep);
  // reconverged: EXEC all-ones again for the WMMAs

  v8f acc = {};
  #pragma unroll
  for (int kt = 0; kt < 5; ++kt) {
    const _Float16* wrow = W3 + rlo * 160 + kt * 32 + hi * 8;
    v16h bf = frag16(wrow, wrow + 16);
    acc = __builtin_amdgcn_wmma_f32_16x16x32_f16(false, afr[kt], false, bf,
                                                 (short)0, acc, false, false);
  }
  int oc = rlo;
  float bias = c3b[oc];
  int pbase = p0 + hi * 8;                // < 108
  #pragma unroll
  for (int r = 0; r < 8; ++r) {
    int pp = pbase + r;                   // < 115
    int carry = (pp >= 100) ? 1 : 0;
    int b2 = b_first + carry;
    int p2 = pp - carry * 100;
    c3o[(b2 * 16 + oc) * 100 + p2] = acc[r] + bias;
  }
}

// ---------------------------------------------------------------------------
// S4 sumpool*scale+bias + act, packed f16 [B][416].
// ---------------------------------------------------------------------------
__global__ void s4_act_pack(const float* __restrict__ c3o, const float* __restrict__ s4w,
                            const float* __restrict__ s4b, _Float16* __restrict__ h2,
                            int B) {
  int id = blockIdx.x * blockDim.x + threadIdx.x;
  if (id >= B * 416) return;
  int b = id / 416, k = id - b * 416;
  _Float16 out = (_Float16)0.f;
  if (k < 400) {
    int ic = k / 25, q = k - ic * 25;
    int y = q / 5, x = q - y * 5;
    const float* base = c3o + (b * 16 + ic) * 100;
    float pool = base[(2 * y) * 10 + 2 * x]     + base[(2 * y) * 10 + 2 * x + 1]
               + base[(2 * y + 1) * 10 + 2 * x] + base[(2 * y + 1) * 10 + 2 * x + 1];
    out = (_Float16)lenet_act(pool * s4w[ic] + s4b[ic]);
  }
  h2[id] = out;
}

// ---------------------------------------------------------------------------
// C5 GEMM on WMMA: [B,416] x [416,128]. Block = one N-tile x 8 M-tiles;
// the 16x416 weight slab (13KB) is staged in LDS once per block,
// B-fragments then come from aligned ds_load_b128.
// ---------------------------------------------------------------------------
__global__ void c5_wmma(const _Float16* __restrict__ h2, const _Float16* __restrict__ W5,
                        const float* __restrict__ c5b, _Float16* __restrict__ c5o,
                        int B) {
  __shared__ alignas(16) _Float16 shB[16 * 416];
  int nt = blockIdx.x & 7;
  int mb = blockIdx.x >> 3;
  for (int c = threadIdx.x; c < 832; c += blockDim.x)
    copy16_g2l(&shB[c * 8], W5 + (size_t)nt * 16 * 416 + (size_t)c * 8);
  async_join();
  __syncthreads();

  int warp = threadIdx.x >> 5;
  int lane = threadIdx.x & 31;
  int mt = mb * 8 + warp;
  if (mt * 16 >= B) return;               // wave-uniform, after barrier
  int hi = lane >> 4, rlo = lane & 15;
  int brow = mt * 16 + rlo;
  int oc = nt * 16 + rlo;
  v8f acc = {};
  #pragma unroll
  for (int kt = 0; kt < 13; ++kt) {
    const _Float16* arow = h2 + (size_t)brow * 416 + kt * 32 + hi * 8;
    v16h a = frag16(arow, arow + 16);
    const _Float16* bp = &shB[rlo * 416 + kt * 32 + hi * 8];
    v16h bf = frag16(bp, bp + 16);
    acc = __builtin_amdgcn_wmma_f32_16x16x32_f16(false, a, false, bf,
                                                 (short)0, acc, false, false);
  }
  float bias = (oc < 120) ? c5b[oc] : 0.f;
  #pragma unroll
  for (int r = 0; r < 8; ++r) {
    int b2 = mt * 16 + hi * 8 + r;
    c5o[b2 * 128 + oc] = (oc < 120) ? (_Float16)(acc[r] + bias) : (_Float16)0.f;
  }
}

// ---------------------------------------------------------------------------
// F6 GEMM on WMMA: [B,128] x [128,96] -> f32 [B][84] (+bias).
// ---------------------------------------------------------------------------
__global__ void f6_wmma(const _Float16* __restrict__ c5o, const _Float16* __restrict__ W6,
                        const float* __restrict__ f6b, float* __restrict__ f6o,
                        int B) {
  int warp = threadIdx.x >> 5;
  int lane = threadIdx.x & 31;
  int w = blockIdx.x * (blockDim.x >> 5) + warp;
  int mt = w / 6, nt = w - mt * 6;
  if (mt * 16 >= B) return;
  int hi = lane >> 4, rlo = lane & 15;
  int brow = mt * 16 + rlo;
  int n = nt * 16 + rlo;
  v8f acc = {};
  #pragma unroll
  for (int kt = 0; kt < 4; ++kt) {
    const _Float16* arow = c5o + (size_t)brow * 128 + kt * 32 + hi * 8;
    v16h a = frag16(arow, arow + 16);
    const _Float16* wrow = W6 + (size_t)n * 128 + kt * 32 + hi * 8;
    v16h bf = frag16(wrow, wrow + 16);
    acc = __builtin_amdgcn_wmma_f32_16x16x32_f16(false, a, false, bf,
                                                 (short)0, acc, false, false);
  }
  if (n < 84) {
    float bias = f6b[n];
    #pragma unroll
    for (int r = 0; r < 8; ++r) {
      int b2 = mt * 16 + hi * 8 + r;
      f6o[b2 * 84 + n] = acc[r] + bias;
    }
  }
}

// ---------------------------------------------------------------------------
// RBF output: out[b][d] = sum_j (f6[b][j] - rbf_w[d][j])^2
// ---------------------------------------------------------------------------
__global__ void rbf_kernel(const float* __restrict__ f6o, const float* __restrict__ rbfw,
                           float* __restrict__ out, int B) {
  int id = blockIdx.x * blockDim.x + threadIdx.x;
  if (id >= B * 10) return;
  int b = id / 10, d = id - b * 10;
  const float* f = f6o + b * 84;
  const float* wv = rbfw + d * 84;
  float s = 0.f;
  #pragma unroll 4
  for (int j = 0; j < 84; ++j) {
    float t = f[j] - wv[j];
    s = fmaf(t, t, s);
  }
  out[id] = s;
}

static inline size_t align256(size_t v) { return (v + 255) & ~(size_t)255; }

extern "C" void kernel_launch(void* const* d_in, const int* in_sizes, int n_in,
                              void* d_out, int out_size, void* d_ws, size_t ws_size,
                              hipStream_t stream) {
  const float* x    = (const float*)d_in[0];
  const float* c1w  = (const float*)d_in[1];
  const float* c1b  = (const float*)d_in[2];
  const float* s2w  = (const float*)d_in[3];
  const float* s2b  = (const float*)d_in[4];
  const float* w3   = (const float*)d_in[5];
  const float* w4   = (const float*)d_in[6];
  const float* w6   = (const float*)d_in[7];
  const float* c3b  = (const float*)d_in[8];
  const float* s4w  = (const float*)d_in[9];
  const float* s4b  = (const float*)d_in[10];
  const float* c5w  = (const float*)d_in[11];
  const float* c5b  = (const float*)d_in[12];
  const float* f6w  = (const float*)d_in[13];
  const float* f6b  = (const float*)d_in[14];
  const float* rbfw = (const float*)d_in[15];
  float* out = (float*)d_out;

  const int B = in_sizes[0] / 1024;   // x is [B,1,32,32]

  char* ws = (char*)d_ws;
  size_t off = 0;
  _Float16* W3  = (_Float16*)(ws + off); off = align256(off + (size_t)16  * 160 * 2);
  _Float16* W5  = (_Float16*)(ws + off); off = align256(off + (size_t)128 * 416 * 2);
  _Float16* W6  = (_Float16*)(ws + off); off = align256(off + (size_t)96  * 128 * 2);
  _Float16* h1  = (_Float16*)(ws + off); off = align256(off + (size_t)B * 6 * 196 * 2);
  float*    c3o = (float*)   (ws + off); off = align256(off + (size_t)B * 1600 * 4);
  _Float16* h2  = (_Float16*)(ws + off); off = align256(off + (size_t)B * 416 * 2);
  _Float16* c5o = (_Float16*)(ws + off); off = align256(off + (size_t)B * 128 * 2);
  float*    f6o = (float*)   (ws + off); off = align256(off + (size_t)B * 84 * 4);
  (void)ws_size;

  const int BLK = 256;

  { // 1) weight prep
    int n = 16 * 160 + 128 * 416 + 96 * 128;
    prep_weights<<<(n + BLK - 1) / BLK, BLK, 0, stream>>>(w3, w4, w6, c5w, f6w, W3, W5, W6);
  }
  { // 2) C1 + S2 + act
    int n = B * 196;
    c1_s2_act<<<(n + BLK - 1) / BLK, BLK, 0, stream>>>(x, c1w, c1b, s2w, s2b, h1, B);
  }
  { // 3) C3 via WMMA (LDS-staged activations, imm-offset LDS gathers)
    int Mtotal = B * 100;
    int blocks = (Mtotal + 127) / 128;
    c3_wmma<<<blocks, BLK, 0, stream>>>(h1, W3, c3b, c3o, Mtotal, B);
  }
  { // 4) S4 + act + pack
    int n = B * 416;
    s4_act_pack<<<(n + BLK - 1) / BLK, BLK, 0, stream>>>(c3o, s4w, s4b, h2, B);
  }
  { // 5) C5 via WMMA (LDS-staged weight slab per N-tile)
    int mb = (B + 127) / 128;
    c5_wmma<<<mb * 8, BLK, 0, stream>>>(h2, W5, c5b, c5o, B);
  }
  { // 6) F6 via WMMA
    int mt = (B + 15) / 16;
    int waves = mt * 6;
    f6_wmma<<<(waves + 7) / 8, BLK, 0, stream>>>(c5o, W6, f6b, f6o, B);
  }
  { // 7) RBF distances
    int n = B * 10;
    rbf_kernel<<<(n + BLK - 1) / BLK, BLK, 0, stream>>>(f6o, rbfw, out, B);
  }
}
